// FullMatchingLayer_19155554140813
// MI455X (gfx1250) — matrix operand order repaired and checked by reading, not compiled, not run on hardware
//
#include <hip/hip_runtime.h>

typedef __attribute__((ext_vector_type(16))) __bf16 v16bf;
typedef __attribute__((ext_vector_type(8)))  __bf16 v8bf;
typedef __attribute__((ext_vector_type(8)))  float  v8f;

namespace {
constexpr int   kB        = 32;
constexpr int   kT        = 512;
constexpr int   kD        = 768;
constexpr int   kP        = 20;
constexpr int   kPpad     = 32;    // P padded to two 16-wide WMMA N tiles
constexpr int   kStr      = 792;   // padded bf16 row stride: 1584B, conflict-free ds_read_b128
constexpr int   kThreads  = 256;   // 8 wave32
constexpr int   kRowsPerWG = 128;  // 8 waves * 16 rows of T
constexpr float kEps      = 1e-12f;
}

union BF16x16 { v16bf v; v8bf h[2]; __bf16 e[16]; };

__global__ __launch_bounds__(kThreads)
void full_matching_wmma(const float* __restrict__ inpA,
                        const float* __restrict__ inpB,
                        const float* __restrict__ Wmat,
                        float* __restrict__ out)
{
    extern __shared__ unsigned char smem[];
    __bf16* sBn   = (__bf16*)smem;                   // [32][kStr] bf16(last*W^2)
    __bf16* sBw   = sBn + kPpad * kStr;              // [32][kStr] bf16(W^2)
    float*  sLast = (float*)(sBw + kPpad * kStr);    // [768] f32 last row of inp_b
    float*  sNl   = sLast + kD;                      // [32] sum(last^2 * W^2)
    float*  sRnl  = sNl + kPpad;                     // [32] 1/sqrt(nl)

    const int tid  = threadIdx.x;
    const int lane = tid & 31;
    const int wave = tid >> 5;
    const int b    = blockIdx.x >> 2;
    const int t0wg = (blockIdx.x & 3) * kRowsPerWG;

    // ---- Phase 0: stage last[b,:], zero nl accumulators ----
    if (tid < kPpad) sNl[tid] = 0.0f;
    for (int d = tid; d < kD; d += kThreads)
        sLast[d] = inpB[((size_t)b * kT + (kT - 1)) * kD + d];
    __syncthreads();

    // ---- Phase 1: build bf16 B operands in LDS + nl reduction ----
    for (int p = 0; p < kPpad; ++p) {
        float partial = 0.0f;
        for (int d = tid; d < kD; d += kThreads) {
            float w2 = 0.0f, bn = 0.0f;
            if (p < kP) {
                float w = Wmat[p * kD + d];
                w2 = w * w;
                float l = sLast[d];
                bn = l * w2;
                partial = fmaf(l * l, w2, partial);
            }
            sBw[p * kStr + d] = (__bf16)w2;
            sBn[p * kStr + d] = (__bf16)bn;
        }
#pragma unroll
        for (int off = 16; off > 0; off >>= 1)
            partial += __shfl_xor(partial, off, 32);
        if (lane == 0 && p < kP) atomicAdd(&sNl[p], partial);   // ds_add_f32
    }
    __syncthreads();
    if (tid < kPpad) sRnl[tid] = rsqrtf(fmaxf(sNl[tid], kEps));
    __syncthreads();

    // ---- Phase 2: fused dual GEMM over K = 768 ----
    // A fragment (16-bit 16x32): lanes 0-15 -> rows, K 0..15; lanes 16-31 -> K 16..31
    const int row  = lane & 15;
    const int kgrp = lane >> 4;
    const int t0   = t0wg + wave * 16;
    const float* Ap = inpA + ((size_t)b * kT + (t0 + row)) * kD;

    // B fragment (16-bit 32x16): lanes 0-15 -> col, K 0..15; lanes 16-31 -> K 16..31
    const __bf16* bn0p = sBn + (size_t)(lane & 15) * kStr;
    const __bf16* bn1p = bn0p + 16 * kStr;
    const __bf16* bw0p = sBw + (size_t)(lane & 15) * kStr;
    const __bf16* bw1p = bw0p + 16 * kStr;

    v8f cn0 = {}, cn1 = {}, cw0 = {}, cw1 = {};

    for (int k0 = 0; k0 < kD; k0 += 32) {
        const int kb = k0 + kgrp * 16;
        float4 x0 = *(const float4*)(Ap + kb);
        float4 x1 = *(const float4*)(Ap + kb + 4);
        float4 x2 = *(const float4*)(Ap + kb + 8);
        float4 x3 = *(const float4*)(Ap + kb + 12);
        if (k0 + 32 < kD) __builtin_prefetch(Ap + kb + 32, 0, 1);  // global_prefetch_b8

        const float av[16] = { x0.x, x0.y, x0.z, x0.w,  x1.x, x1.y, x1.z, x1.w,
                               x2.x, x2.y, x2.z, x2.w,  x3.x, x3.y, x3.z, x3.w };
        BF16x16 fa, fs;
#pragma unroll
        for (int j = 0; j < 16; ++j) {
            fa.e[j] = (__bf16)av[j];
            fs.e[j] = (__bf16)(av[j] * av[j]);
        }

        const int kr = k0 + kgrp * 16;
        BF16x16 fbn0, fbn1, fbw0, fbw1;
        fbn0.h[0] = *(const v8bf*)(bn0p + kr);  fbn0.h[1] = *(const v8bf*)(bn0p + kr + 8);
        fbn1.h[0] = *(const v8bf*)(bn1p + kr);  fbn1.h[1] = *(const v8bf*)(bn1p + kr + 8);
        fbw0.h[0] = *(const v8bf*)(bw0p + kr);  fbw0.h[1] = *(const v8bf*)(bw0p + kr + 8);
        fbw1.h[0] = *(const v8bf*)(bw1p + kr);  fbw1.h[1] = *(const v8bf*)(bw1p + kr + 8);

        cn0 = __builtin_amdgcn_wmma_f32_16x16x32_bf16(false, fa.v, false, fbn0.v, (short)0, cn0, false, false);
        cn1 = __builtin_amdgcn_wmma_f32_16x16x32_bf16(false, fa.v, false, fbn1.v, (short)0, cn1, false, false);
        cw0 = __builtin_amdgcn_wmma_f32_16x16x32_bf16(false, fs.v, false, fbw0.v, (short)0, cw0, false, false);
        cw1 = __builtin_amdgcn_wmma_f32_16x16x32_bf16(false, fs.v, false, fbw1.v, (short)0, cw1, false, false);
    }

    // ---- Phase 3: normalize + store both output copies ----
    // C layout: VGPR i -> M = i + 8*(lane>=16), N = lane&15
    float* out0 = out;
    float* out1 = out + (size_t)kB * kT * kP;
    const int   pc = lane & 15;
    const float r0 = sRnl[pc];
    const float r1 = sRnl[pc + 16];
#pragma unroll
    for (int i = 0; i < 8; ++i) {
        const int    t    = t0 + i + 8 * kgrp;
        const size_t base = ((size_t)b * kT + t) * kP;
        const float  y0   = cn0[i] * rsqrtf(fmaxf(cw0[i], kEps)) * r0;
        out0[base + pc] = y0;
        out1[base + pc] = y0;
        if (pc < kP - 16) {   // only p = 16..19 valid in the high N tile
            const float y1 = cn1[i] * rsqrtf(fmaxf(cw1[i], kEps)) * r1;
            out0[base + 16 + pc] = y1;
            out1[base + 16 + pc] = y1;
        }
    }
}

extern "C" void kernel_launch(void* const* d_in, const int* in_sizes, int n_in,
                              void* d_out, int out_size, void* d_ws, size_t ws_size,
                              hipStream_t stream) {
    (void)in_sizes; (void)n_in; (void)out_size; (void)d_ws; (void)ws_size;
    const float* inpA = (const float*)d_in[0];
    const float* inpB = (const float*)d_in[1];
    const float* Wmat = (const float*)d_in[2];
    float* out = (float*)d_out;

    const size_t shmem = (size_t)2 * kPpad * kStr * sizeof(__bf16)
                       + (size_t)(kD + 2 * kPpad) * sizeof(float);
    dim3 grid(kB * (kT / kRowsPerWG));   // 128 workgroups
    dim3 block(kThreads);                // 8 wave32
    full_matching_wmma<<<grid, block, shmem, stream>>>(inpA, inpB, Wmat, out);
}
